// Attention_16904991277120
// MI455X (gfx1250) — compile-verified, matrix operand order checked by reading, not run on hardware
//
#include <hip/hip_runtime.h>
#include <hip/hip_bf16.h>

// ---------------------------------------------------------------------------
// Shapes (fixed by the reference): B=64, T=1024, N=1024
// d_out layout: c_t [B*N] | attn [B*T] | coverage_new [B*T]  (all fp32)
// d_ws layout:  W_h bf16 [N*N ushort] | scores [B*T f32] | dec_fea [B*N f32]
//               | ct_partials [8*B*N f32]
// ---------------------------------------------------------------------------

#define BB 64
#define TT 1024
#define NN 1024
#define TM 64            // T-rows per workgroup in the scores kernel
#define LDSTRIDE 1032    // padded LDS row stride in bf16 elems (bank-conflict-free)

typedef __attribute__((ext_vector_type(16))) __bf16 v16bf;
typedef __attribute__((ext_vector_type(8)))  __bf16 v8bf;
typedef __attribute__((ext_vector_type(8)))  float  v8f;

__device__ __forceinline__ unsigned short f32_to_bf16u(float f) {
  union { float f; unsigned u; } v; v.f = f;
  unsigned b = v.u;
  unsigned r = (b + 0x7FFFu + ((b >> 16) & 1u)) >> 16;   // round-to-nearest-even
  return (unsigned short)r;
}

// Load one 16x32 (or 32x16) bf16 WMMA fragment.  p already points at this
// lane's first 8-element chunk (row base + k0 + half*8); second chunk is +16.
__device__ __forceinline__ v16bf load_frag(const unsigned short* p) {
  v8bf lo = *reinterpret_cast<const v8bf*>(p);
  v8bf hi = *reinterpret_cast<const v8bf*>(p + 16);
  v16bf f;
#pragma unroll
  for (int i = 0; i < 8; ++i) { f[i] = lo[i]; f[i + 8] = hi[i]; }
  return f;
}

__device__ __forceinline__ float fast_tanh(float x) {
  x = fminf(fmaxf(x, -15.0f), 15.0f);
  float e = __expf(2.0f * x);
  return (e - 1.0f) / (e + 1.0f);
}

// --------------------------- K0: W_h fp32 -> bf16 ---------------------------
__global__ __launch_bounds__(256) void k_cvt_wh(const float* __restrict__ wh,
                                                unsigned short* __restrict__ whbf) {
  int idx = (blockIdx.x * 256 + threadIdx.x) * 4;
  float4 f = *reinterpret_cast<const float4*>(wh + idx);
  whbf[idx + 0] = f32_to_bf16u(f.x);
  whbf[idx + 1] = f32_to_bf16u(f.y);
  whbf[idx + 2] = f32_to_bf16u(f.z);
  whbf[idx + 3] = f32_to_bf16u(f.w);
}

// ------------------ K1: dec_fea = s_t_hat @ dec_W^T + dec_b -----------------
__global__ __launch_bounds__(256) void k_decfea(const float* __restrict__ s_t_hat,
                                                const float* __restrict__ dec_W,
                                                const float* __restrict__ dec_b,
                                                float* __restrict__ dec_fea) {
  __shared__ float s_row[NN];
  const int b = blockIdx.y;
  const int m = blockIdx.x * 256 + threadIdx.x;
  for (int j = threadIdx.x; j < NN; j += 256) s_row[j] = s_t_hat[(size_t)b * NN + j];
  __syncthreads();
  const float* wrow = dec_W + (size_t)m * NN;
  float acc = dec_b[m];
  for (int k = 0; k < NN; k += 4) {
    float4 w = *reinterpret_cast<const float4*>(wrow + k);
    acc += w.x * s_row[k] + w.y * s_row[k + 1] + w.z * s_row[k + 2] + w.w * s_row[k + 3];
  }
  dec_fea[(size_t)b * NN + m] = acc;
}

// ---- K2: scores[b,t] = sum_n v[n]*tanh( (h W_h^T)[t,n] + dec_fea[b,n]
//                                          + cov[b,t]*W_c[n] ) via bf16 WMMA ----
__global__ __launch_bounds__(256) void k_scores(const float* __restrict__ h,
                                                const float* __restrict__ cov,
                                                const unsigned short* __restrict__ whbf,
                                                const float* __restrict__ dec_fea,
                                                const float* __restrict__ Wc,
                                                const float* __restrict__ vv,
                                                float* __restrict__ scores) {
  extern __shared__ char smem[];
  unsigned short* sh_h = reinterpret_cast<unsigned short*>(smem);           // TM x LDSTRIDE bf16
  float* sh_dec  = reinterpret_cast<float*>(smem + TM * LDSTRIDE * 2);      // [NN]
  float* sh_v    = sh_dec + NN;                                             // [NN]
  float* sh_wc   = sh_v + NN;                                               // [NN]
  float* sh_cov  = sh_wc + NN;                                              // [TM]
  float* sh_part = sh_cov + TM;                                             // [8][TM]

  const int tid = threadIdx.x;
  const int b   = blockIdx.y;
  const int t0  = blockIdx.x * TM;

  // Stage h tile (fp32 -> bf16) into LDS, coalesced float4 loads.
  const float* hbase = h + ((size_t)b * TT + t0) * NN;
#pragma unroll 2
  for (int it = 0; it < (TM * NN / 4) / 256; ++it) {
    int idx = it * 256 + tid;               // float4 slot
    int r = idx >> 8;                       // 256 float4 per row
    int k = (idx & 255) * 4;
    float4 f = *reinterpret_cast<const float4*>(hbase + (size_t)r * NN + k);
    unsigned u0 = (unsigned)f32_to_bf16u(f.x) | ((unsigned)f32_to_bf16u(f.y) << 16);
    unsigned u1 = (unsigned)f32_to_bf16u(f.z) | ((unsigned)f32_to_bf16u(f.w) << 16);
    *reinterpret_cast<uint2*>(&sh_h[r * LDSTRIDE + k]) = make_uint2(u0, u1);
  }
  for (int j = tid; j < NN; j += 256) {
    sh_dec[j] = dec_fea[(size_t)b * NN + j];
    sh_v[j]   = vv[j];
    sh_wc[j]  = Wc[j];
  }
  if (tid < TM) sh_cov[tid] = cov[(size_t)b * TT + t0 + tid];
  __syncthreads();

  const int lane  = tid & 31;
  const int wave  = tid >> 5;
  const int half  = lane >> 4;              // which 16-lane group
  const int l16   = lane & 15;
  const int nwave = wave * 128;             // this wave's 128 N-columns

  float p[4][8];
#pragma unroll
  for (int mt = 0; mt < 4; ++mt)
#pragma unroll
    for (int r = 0; r < 8; ++r) p[mt][r] = 0.0f;

  const unsigned short* arow = sh_h + l16 * LDSTRIDE + half * 8;

  for (int ntb = 0; ntb < 4; ++ntb) {       // 2 N-tiles per block
    const int nA = nwave + ntb * 32 + l16;
    const int nB = nA + 16;
    const float vnA = sh_v[nA], dfA = sh_dec[nA], wcA = sh_wc[nA];
    const float vnB = sh_v[nB], dfB = sh_dec[nB], wcB = sh_wc[nB];
    const unsigned short* browA = whbf + (size_t)nA * NN + half * 8;
    const unsigned short* browB = whbf + (size_t)nB * NN + half * 8;

    v8f acc[4][2];
#pragma unroll
    for (int mt = 0; mt < 4; ++mt) { acc[mt][0] = {}; acc[mt][1] = {}; }

#pragma unroll 2
    for (int ks = 0; ks < 32; ++ks) {       // K = 1024 in steps of 32
      const int k0 = ks * 32;
      v16bf fbA = load_frag(browA + k0);    // W_h rows nA/nB, L2-resident
      v16bf fbB = load_frag(browB + k0);
#pragma unroll
      for (int mt = 0; mt < 4; ++mt) {
        v16bf fa = load_frag(arow + mt * 16 * LDSTRIDE + k0);
        acc[mt][0] = __builtin_amdgcn_wmma_f32_16x16x32_bf16(
            false, fa, false, fbA, (short)0, acc[mt][0], false, false);
        acc[mt][1] = __builtin_amdgcn_wmma_f32_16x16x32_bf16(
            false, fa, false, fbB, (short)0, acc[mt][1], false, false);
      }
    }

    // Epilogue: + dec_fea + cov*W_c, tanh, weight by v, accumulate per row.
    // C/D layout: acc[mt][j][r] is (m = mt*16 + half*8 + r, n = n0 + l16).
#pragma unroll
    for (int mt = 0; mt < 4; ++mt) {
#pragma unroll
      for (int r = 0; r < 8; ++r) {
        float c  = sh_cov[mt * 16 + half * 8 + r];
        float x0 = acc[mt][0][r] + dfA + c * wcA;
        float x1 = acc[mt][1][r] + dfB + c * wcB;
        p[mt][r] += vnA * fast_tanh(x0) + vnB * fast_tanh(x1);
      }
    }
  }

  // Deterministic reduction over the 16 lanes holding each row's partials.
#pragma unroll
  for (int mt = 0; mt < 4; ++mt)
#pragma unroll
    for (int r = 0; r < 8; ++r) {
      float s = p[mt][r];
      s += __shfl_xor(s, 1, 32);
      s += __shfl_xor(s, 2, 32);
      s += __shfl_xor(s, 4, 32);
      s += __shfl_xor(s, 8, 32);
      p[mt][r] = s;
    }
  if (l16 == 0) {
#pragma unroll
    for (int mt = 0; mt < 4; ++mt)
#pragma unroll
      for (int r = 0; r < 8; ++r)
        sh_part[wave * TM + mt * 16 + half * 8 + r] = p[mt][r];
  }
  __syncthreads();
  if (tid < TM) {
    float s = 0.0f;
#pragma unroll
    for (int w = 0; w < 8; ++w) s += sh_part[w * TM + tid];
    scores[(size_t)b * TT + t0 + tid] = s;
  }
}

// ---------------- K3: masked softmax + renorm, attn & coverage_new ----------
__global__ __launch_bounds__(256) void k_softmax(const float* __restrict__ scores,
                                                 const float* __restrict__ mask,
                                                 const float* __restrict__ cov,
                                                 float* __restrict__ out) {
  __shared__ float red[256];
  const int b = blockIdx.x;
  const int tid = threadIdx.x;
  float s[4];
  float mx = -1e30f;
#pragma unroll
  for (int j = 0; j < 4; ++j) {
    s[j] = scores[(size_t)b * TT + tid + 256 * j];
    mx = fmaxf(mx, s[j]);
  }
  red[tid] = mx; __syncthreads();
  for (int off = 128; off > 0; off >>= 1) {
    if (tid < off) red[tid] = fmaxf(red[tid], red[tid + off]);
    __syncthreads();
  }
  mx = red[0]; __syncthreads();

  float e[4]; float sum = 0.0f;
#pragma unroll
  for (int j = 0; j < 4; ++j) {
    float m = mask[(size_t)b * TT + tid + 256 * j];
    e[j] = __expf(s[j] - mx) * m;     // masked exp; attn = e / sum(e)
    sum += e[j];
  }
  red[tid] = sum; __syncthreads();
  for (int off = 128; off > 0; off >>= 1) {
    if (tid < off) red[tid] += red[tid + off];
    __syncthreads();
  }
  const float inv = 1.0f / red[0];

  float* attn_out = out + (size_t)BB * NN;
  float* cov_out  = attn_out + (size_t)BB * TT;
#pragma unroll
  for (int j = 0; j < 4; ++j) {
    int t = tid + 256 * j;
    float a = e[j] * inv;
    attn_out[(size_t)b * TT + t] = a;
    cov_out[(size_t)b * TT + t]  = cov[(size_t)b * TT + t] + a;
  }
}

// ----------- K4: c_t partials over t-chunks; K5: deterministic reduce -------
__global__ __launch_bounds__(256) void k_ctx_part(const float* __restrict__ h,
                                                  const float* __restrict__ attn,
                                                  float* __restrict__ ct_part) {
  const int b = blockIdx.y;
  const int chunk = blockIdx.x;             // 0..7, 128 t each
  const int n4 = threadIdx.x * 4;
  float4 acc = make_float4(0.f, 0.f, 0.f, 0.f);
  const int tstart = chunk * 128;
  for (int t = tstart; t < tstart + 128; ++t) {
    float a = attn[(size_t)b * TT + t];
    float4 hv = *reinterpret_cast<const float4*>(h + ((size_t)b * TT + t) * NN + n4);
    acc.x += a * hv.x; acc.y += a * hv.y; acc.z += a * hv.z; acc.w += a * hv.w;
  }
  *reinterpret_cast<float4*>(ct_part + ((size_t)chunk * BB + b) * NN + n4) = acc;
}

__global__ __launch_bounds__(256) void k_ctx_reduce(const float* __restrict__ ct_part,
                                                    float* __restrict__ ct) {
  const int idx = blockIdx.x * 256 + threadIdx.x;   // b*NN + n
  float s = 0.0f;
#pragma unroll
  for (int c = 0; c < 8; ++c) s += ct_part[(size_t)c * BB * NN + idx];
  ct[idx] = s;
}

// ---------------------------------------------------------------------------
extern "C" void kernel_launch(void* const* d_in, const int* in_sizes, int n_in,
                              void* d_out, int out_size, void* d_ws, size_t ws_size,
                              hipStream_t stream) {
  const float* s_t_hat = (const float*)d_in[0];
  const float* h       = (const float*)d_in[1];
  const float* maskp   = (const float*)d_in[2];
  const float* cov     = (const float*)d_in[3];
  const float* W_h     = (const float*)d_in[4];
  const float* W_c     = (const float*)d_in[5];
  const float* dec_W   = (const float*)d_in[6];
  const float* dec_b   = (const float*)d_in[7];
  const float* v       = (const float*)d_in[8];
  float* out = (float*)d_out;

  // Workspace carve-up (~4.5 MB total)
  unsigned short* whbf = (unsigned short*)d_ws;                               // 2 MB
  float* scores  = (float*)((char*)d_ws + (size_t)NN * NN * sizeof(unsigned short));
  float* dec_fea = scores + (size_t)BB * TT;                                  // 256 KB
  float* ct_part = dec_fea + (size_t)BB * NN;                                 // 2 MB

  // K0: W_h -> bf16 (L2-resident weight copy)
  k_cvt_wh<<<(NN * NN) / 1024, 256, 0, stream>>>(W_h, whbf);

  // K1: dec_fea = s_t_hat @ dec_W^T + dec_b
  k_decfea<<<dim3(NN / 256, BB), 256, 0, stream>>>(s_t_hat, dec_W, dec_b, dec_fea);

  // K2: fused GEMM + tanh + v-dot -> scores  (dynamic LDS: h tile + vectors)
  const size_t smem = (size_t)TM * LDSTRIDE * 2      // h tile (bf16)
                    + 3 * NN * sizeof(float)          // dec_fea row, v, W_c
                    + TM * sizeof(float)              // coverage tile
                    + 8 * TM * sizeof(float);         // per-wave partials
  k_scores<<<dim3(TT / TM, BB), 256, smem, stream>>>(h, cov, whbf, dec_fea, W_c, v,
                                                     scores);

  // K3: masked softmax + renorm -> attn, coverage_new
  k_softmax<<<BB, 256, 0, stream>>>(scores, maskp, cov, out);

  // K4/K5: c_t = attn . h  (fp32, coalesced, deterministic two-phase reduce)
  const float* attn = out + (size_t)BB * NN;
  k_ctx_part<<<dim3(8, BB), 256, 0, stream>>>(h, attn, ct_part);
  k_ctx_reduce<<<(BB * NN) / 256, 256, 0, stream>>>(ct_part, out);
}
